// TopKAttentionWrapper_9028021256288
// MI455X (gfx1250) — matrix operand order repaired and checked by reading, not compiled, not run on hardware
//
#include <hip/hip_runtime.h>
#include <hip/hip_bf16.h>
#include <stdint.h>

// ---------------------------------------------------------------------------
// Top-k sparse attention for MI455X (gfx1250), bf16 WMMA pipeline with
// TDM (tensor_load_to_lds) weight-panel staging.
//   B=4, N=1024, C=768, H=12, Dh=64, K_TOP=16
// d_out = [ x_out (4096*768 f32) | topk_attn (4*12*1024*1024 f32) ]
// ---------------------------------------------------------------------------

#define BATCH   4
#define SEQ     1024
#define CDIM    768
#define C3      (3 * CDIM)
#define NHEADS  12
#define DH      64
#define ROWS    (BATCH * SEQ)        // 4096
#define KTOP    16
#define XOUT_ELEMS ((size_t)ROWS * CDIM)

typedef __attribute__((ext_vector_type(16))) __bf16   v16bf;
typedef __attribute__((ext_vector_type(8)))  float    v8f;
typedef __attribute__((ext_vector_type(4)))  uint32_t u32x4;
typedef __attribute__((ext_vector_type(8)))  uint32_t u32x8;

union BfFrag { v16bf v; int4 q[2]; };

__device__ __forceinline__ uint16_t f2bf(float f) {
  uint32_t u = __float_as_uint(f);
  uint32_t r = u + 0x7FFFu + ((u >> 16) & 1u);   // round-to-nearest-even
  return (uint16_t)(r >> 16);
}

// A-matrix 16x32 bf16 fragment (ISA 7.12.2): lane L<16 row M=L holds
// K=[0..7] in v0..3 and K=[16..23] in v4..7; lane L+16 holds K=[8..15],[24..31].
__device__ __forceinline__ v16bf load_a_frag(const uint16_t* A, int lda,
                                             int row0, int k0, int half, int r) {
  BfFrag f;
  const uint16_t* p = A + (size_t)(row0 + r) * lda + k0 + 8 * half;
  f.q[0] = *(const int4*)(p);
  f.q[1] = *(const int4*)(p + 16);
  return f.v;
}

// B-matrix 32x16 bf16 fragment: lane (half h, n=r) holds K=[8h..8h+7] and
// K=[16+8h..16+8h+7] of column n.  Source stored row-major as Bm[n][k].
__device__ __forceinline__ v16bf load_b_frag(const uint16_t* Bm, int ldb,
                                             int n0, int k0, int half, int r) {
  BfFrag f;
  const uint16_t* p = Bm + (size_t)(n0 + r) * ldb + k0 + 8 * half;
  f.q[0] = *(const int4*)(p);
  f.q[1] = *(const int4*)(p + 16);
  return f.v;
}

// ---------------------------------------------------------------------------
// TDM: DMA a [64 x 768] bf16 weight panel (row-major, contiguous rows) into
// LDS.  D# per CDNA5 ISA, 2-group (<=2D) form.
// Issued by one wave; EXEC is ignored by TDM.
// ---------------------------------------------------------------------------
__device__ __forceinline__ void tdm_load_panel(const uint16_t* gsrc,
                                               uint16_t* lds_dst,
                                               uint32_t tensor_rows) {
  uint64_t ga  = (uint64_t)(uintptr_t)gsrc;
  uint32_t lds = (uint32_t)(uintptr_t)lds_dst;   // flat addr[31:0] == LDS offset
  u32x4 g0;
  g0[0] = 1u;                                    // count=1, user descriptor
  g0[1] = lds;                                   // lds_addr
  g0[2] = (uint32_t)ga;                          // global_addr[31:0]
  g0[3] = (uint32_t)((ga >> 32) & 0x1FFFFFFull)  // global_addr[56:32]
        | (2u << 30);                            // type=2 ("image")
  u32x8 g1;
  g1[0] = 1u << 16;                              // data_size=1 (2 bytes)
  g1[1] = (uint32_t)CDIM << 16;                  // tensor_dim0[15:0]=768
  g1[2] = tensor_rows << 16;                     // tensor_dim1[15:0]
  g1[3] = (uint32_t)CDIM << 16;                  // tile_dim0=768
  g1[4] = 64u;                                   // tile_dim1=64
  g1[5] = (uint32_t)CDIM;                        // tensor_dim0_stride=768
  g1[6] = 0u;
  g1[7] = 0u;
  asm volatile("tensor_load_to_lds %0, %1" :: "s"(g0), "s"(g1) : "memory");
  __builtin_amdgcn_s_wait_tensorcnt(0);
}

// One wave: 16x64 output tile, A from global (double-buffered), B from LDS.
__device__ __forceinline__ void wave_gemm_16x64_lds(const uint16_t* A, int lda,
                                                    const uint16_t* sB,  // [64][768]
                                                    int row0, int K,
                                                    v8f acc[4], int half, int r) {
  v16bf a_cur = load_a_frag(A, lda, row0, 0, half, r);
  for (int k0 = 0; k0 < K; k0 += 32) {
    v16bf a_nxt = a_cur;
    if (k0 + 32 < K) {
      __builtin_prefetch(A + (size_t)(row0 + r) * lda + k0 + 64, 0, 1);
      a_nxt = load_a_frag(A, lda, row0, k0 + 32, half, r);
    }
#pragma unroll
    for (int t = 0; t < 4; ++t) {
      v16bf b = load_b_frag(sB, CDIM, 16 * t, k0, half, r);  // ds_load_b128 x2
      acc[t] = __builtin_amdgcn_wmma_f32_16x16x32_bf16(
          false, a_cur, false, b, (short)0, acc[t], false, false);
    }
    a_cur = a_nxt;
  }
}

// ---------------------------------------------------------------------------
// fp32 -> bf16 convert
// ---------------------------------------------------------------------------
__global__ __launch_bounds__(256) void cvt_bf16_kernel(const float* __restrict__ in,
                                                       uint16_t* __restrict__ out,
                                                       int n) {
  int i = blockIdx.x * 256 + threadIdx.x;
  if (i < n) out[i] = f2bf(in[i]);
}

// ---------------------------------------------------------------------------
// QKV GEMM: block = 128 rows x 64 cols; weight panel staged via TDM.
// [4096 x 768] @ [2304 x 768]^T + bias -> Q/K bf16, V f32, layout [B,H,N,Dh]
// ---------------------------------------------------------------------------
__global__ __launch_bounds__(256) void qkv_gemm_kernel(
    const uint16_t* __restrict__ Xbf, const uint16_t* __restrict__ Wbf,
    const float* __restrict__ bias, uint16_t* __restrict__ Qbf,
    uint16_t* __restrict__ Kbf, float* __restrict__ Vf) {
  __shared__ __align__(128) uint16_t s_B[64 * CDIM];   // 96 KB weight panel

  const int lane = threadIdx.x & 31;
  const int wave = threadIdx.x >> 5;
  const int half = lane >> 4, r = lane & 15;
  const int row0 = (blockIdx.x % (ROWS / 128)) * 128 + wave * 16;
  const int col0 = (blockIdx.x / (ROWS / 128)) * 64;

  if (wave == 0)
    tdm_load_panel(Wbf + (size_t)col0 * CDIM, s_B, C3);
  __syncthreads();

  v8f acc[4] = {};
  wave_gemm_16x64_lds(Xbf, CDIM, s_B, row0, CDIM, acc, half, r);

#pragma unroll
  for (int t = 0; t < 4; ++t) {
    const int col  = col0 + 16 * t + r;
    const float bc = bias[col];
    const int sec  = col / CDIM;                 // 0=Q 1=K 2=V (uniform per t)
    const int win  = col % CDIM;
    const int hh   = win / DH, d = win % DH;
#pragma unroll
    for (int g = 0; g < 8; ++g) {
      const int m = row0 + 8 * half + g;         // global row
      const int b = m >> 10, qq = m & (SEQ - 1);
      const float val = acc[t][g] + bc;
      const size_t idx = ((size_t)((b * NHEADS + hh) * SEQ + qq)) * DH + d;
      if (sec == 0)      Qbf[idx] = f2bf(val);
      else if (sec == 1) Kbf[idx] = f2bf(val);
      else               Vf[idx]  = val;
    }
  }
}

// ---------------------------------------------------------------------------
// Attention: scores via WMMA, top-16 select, renormalize (= softmax over the
// 16 selected logits), dense topk_attn write, sparse V aggregation.
// One block = one (b,h) and 16 query rows.  8 waves.
// ---------------------------------------------------------------------------
__global__ __launch_bounds__(256) void attn_topk_kernel(
    const uint16_t* __restrict__ Qbf, const uint16_t* __restrict__ Kbf,
    const float* __restrict__ Vf, float* __restrict__ topk_out,
    uint16_t* __restrict__ Hbf) {
  __shared__ float s_scores[16][SEQ];            // 64 KB
  __shared__ float s_topv[16][KTOP];
  __shared__ int   s_topi[16][KTOP];

  const int lane = threadIdx.x & 31;
  const int wave = threadIdx.x >> 5;
  const int half = lane >> 4, r = lane & 15;
  const int qt = blockIdx.x & 63;                // query tile within head
  const int bh = blockIdx.x >> 6;                // b*12 + h
  const int q0 = qt * 16;

  const uint16_t* Qh = Qbf + (size_t)bh * SEQ * DH;
  const uint16_t* Kh = Kbf + (size_t)bh * SEQ * DH;
  const float*    Vh = Vf  + (size_t)bh * SEQ * DH;

  // ---- phase 1: scaled scores S[16 x 1024] for this wave's 128 keys ----
  for (int kt = 0; kt < 8; ++kt) {
    const int key0 = wave * 128 + kt * 16;
    v8f acc = {};
#pragma unroll
    for (int k0 = 0; k0 < DH; k0 += 32) {
      v16bf a = load_a_frag(Qh, DH, q0, k0, half, r);
      v16bf b = load_b_frag(Kh, DH, key0, k0, half, r);
      acc = __builtin_amdgcn_wmma_f32_16x16x32_bf16(
          false, a, false, b, (short)0, acc, false, false);
    }
#pragma unroll
    for (int g = 0; g < 8; ++g)
      s_scores[8 * half + g][key0 + r] = acc[g] * 0.125f;   // * Dh^-0.5
  }
  __syncthreads();

  // ---- phase 2/3: per-row top-16, softmax over survivors, dense rebuild ----
  for (int rr = 0; rr < 2; ++rr) {
    const int row = wave + rr * 8;
    for (int it = 0; it < KTOP; ++it) {
      float bestv = -3.0e38f; int besti = SEQ;
      for (int i = 0; i < 32; ++i) {
        const int idx = lane + (i << 5);
        const float v = s_scores[row][idx];
        if (v > bestv) { bestv = v; besti = idx; }
      }
#pragma unroll
      for (int m = 16; m >= 1; m >>= 1) {
        const float ov = __shfl_xor(bestv, m, 32);
        const int   oi = __shfl_xor(besti, m, 32);
        if (ov > bestv || (ov == bestv && oi < besti)) { bestv = ov; besti = oi; }
      }
      if (lane == 0) {
        s_topv[row][it] = bestv;
        s_topi[row][it] = besti;
        s_scores[row][besti] = -3.4e38f;         // mask out for next pass
      }
    }
    // softmax over the 16 selected logits (== reference's renormalized top-k)
    const float v0 = s_topv[row][0];
    float e = (lane < KTOP) ? __expf(s_topv[row][lane] - v0) : 0.f;
    float den = e;
#pragma unroll
    for (int m = 16; m >= 1; m >>= 1) den += __shfl_xor(den, m, 32);
    const float wgt = e / fmaxf(den, 1e-6f);
    // zero the row, scatter normalized weights (wave-private, DS in-order)
    for (int i = 0; i < 32; ++i) s_scores[row][lane + (i << 5)] = 0.f;
    if (lane < KTOP) {
      s_topv[row][lane] = wgt;
      s_scores[row][s_topi[row][lane]] = wgt;
    }
  }
  __syncthreads();

  // ---- phase 4: stream dense topk_attn rows, coalesced ----
  const size_t obase = ((size_t)bh * SEQ + q0) * SEQ;
  for (int i = 0; i < 64; ++i) {
    const int flat = threadIdx.x + (i << 8);
    const int row = flat >> 10, col = flat & (SEQ - 1);
    topk_out[obase + (size_t)row * SEQ + col] = s_scores[row][col];
  }

  // ---- phase 5: head_out[q][d] = sum_j w_j * V[key_j][d] ----
  const int b = bh / NHEADS, h = bh % NHEADS;
  for (int rr = 0; rr < 2; ++rr) {
    const int row = wave + rr * 8;
    float a0 = 0.f, a1 = 0.f;
#pragma unroll
    for (int j = 0; j < KTOP; ++j) {
      const int key = s_topi[row][j];
      const float wgt = s_topv[row][j];
      const float* vp = Vh + (size_t)key * DH;
      a0 += wgt * vp[lane];
      a1 += wgt * vp[lane + 32];
    }
    const size_t hb = ((size_t)(b * SEQ + q0 + row)) * CDIM + h * DH;
    Hbf[hb + lane]      = f2bf(a0);
    Hbf[hb + lane + 32] = f2bf(a1);
  }
}

// ---------------------------------------------------------------------------
// Output projection: block = 128 rows x 64 cols, TDM-staged weight panel.
// [4096 x 768] @ [768 x 768]^T + bias -> d_out (f32)
// ---------------------------------------------------------------------------
__global__ __launch_bounds__(256) void out_proj_kernel(
    const uint16_t* __restrict__ Hbf, const uint16_t* __restrict__ Wobf,
    const float* __restrict__ bias, float* __restrict__ out) {
  __shared__ __align__(128) uint16_t s_B[64 * CDIM];   // 96 KB weight panel

  const int lane = threadIdx.x & 31;
  const int wave = threadIdx.x >> 5;
  const int half = lane >> 4, r = lane & 15;
  const int row0 = (blockIdx.x % (ROWS / 128)) * 128 + wave * 16;
  const int col0 = (blockIdx.x / (ROWS / 128)) * 64;

  if (wave == 0)
    tdm_load_panel(Wobf + (size_t)col0 * CDIM, s_B, CDIM);
  __syncthreads();

  v8f acc[4] = {};
  wave_gemm_16x64_lds(Hbf, CDIM, s_B, row0, CDIM, acc, half, r);

#pragma unroll
  for (int t = 0; t < 4; ++t) {
    const int col = col0 + 16 * t + r;
    const float bc = bias[col];
#pragma unroll
    for (int g = 0; g < 8; ++g) {
      const int m = row0 + 8 * half + g;
      out[(size_t)m * CDIM + col] = acc[t][g] + bc;
    }
  }
}

// ---------------------------------------------------------------------------
extern "C" void kernel_launch(void* const* d_in, const int* in_sizes, int n_in,
                              void* d_out, int out_size, void* d_ws, size_t ws_size,
                              hipStream_t stream) {
  const float* x     = (const float*)d_in[0];
  const float* w_in  = (const float*)d_in[1];
  const float* b_in  = (const float*)d_in[2];
  const float* w_out = (const float*)d_in[3];
  const float* b_out = (const float*)d_in[4];

  char* ws = (char*)d_ws;
  size_t off = 0;
  auto alloc = [&](size_t bytes) {
    char* p = ws + off;
    off = (off + bytes + 255) & ~(size_t)255;
    return p;
  };
  uint16_t* Xbf   = (uint16_t*)alloc((size_t)ROWS * CDIM * 2);
  uint16_t* Winbf = (uint16_t*)alloc((size_t)C3 * CDIM * 2);
  uint16_t* Wobf  = (uint16_t*)alloc((size_t)CDIM * CDIM * 2);
  uint16_t* Qbf   = (uint16_t*)alloc((size_t)BATCH * NHEADS * SEQ * DH * 2);
  uint16_t* Kbf   = (uint16_t*)alloc((size_t)BATCH * NHEADS * SEQ * DH * 2);
  float*    Vf    = (float*)   alloc((size_t)BATCH * NHEADS * SEQ * DH * 4);
  uint16_t* Hbf   = (uint16_t*)alloc((size_t)ROWS * CDIM * 2);

  const int nX = ROWS * CDIM, nWi = C3 * CDIM, nWo = CDIM * CDIM;
  cvt_bf16_kernel<<<(nX  + 255) / 256, 256, 0, stream>>>(x,     Xbf,   nX);
  cvt_bf16_kernel<<<(nWi + 255) / 256, 256, 0, stream>>>(w_in,  Winbf, nWi);
  cvt_bf16_kernel<<<(nWo + 255) / 256, 256, 0, stream>>>(w_out, Wobf,  nWo);

  qkv_gemm_kernel<<<(ROWS / 128) * (C3 / 64), 256, 0, stream>>>(
      Xbf, Winbf, b_in, Qbf, Kbf, Vf);

  float* topk_out = (float*)d_out + XOUT_ELEMS;
  attn_topk_kernel<<<BATCH * NHEADS * (SEQ / 16), 256, 0, stream>>>(
      Qbf, Kbf, Vf, topk_out, Hbf);

  out_proj_kernel<<<(ROWS / 128) * (CDIM / 64), 256, 0, stream>>>(
      Hbf, Wobf, b_out, (float*)d_out);
}
